// SparseTransformerBlock_45303315038805
// MI455X (gfx1250) — matrix operand order repaired and checked by reading, not compiled, not run on hardware
//
#include <hip/hip_runtime.h>
#include <hip/hip_bf16.h>

// ---------------------------------------------------------------------------
// SparseTransformerBlock for MI455X (gfx1250), bf16 WMMA pipeline.
// N=4096 tokens, C=1024, H=16 heads, D=64, FFN M=4096. fp32 in/out.
// All GEMMs + attention via v_wmma_f32_16x16x32_bf16 (fp32 accum).
// Flash attention is a double-buffered pipeline: K tiles DMA'd by the Tensor
// Data Mover (tensor_load_to_lds + s_wait_tensorcnt) one tile ahead of the
// compute, V tiles staged transposed in LDS so P*V B-fragments are contiguous
// ds_load_b128 reads. One workgroup barrier per K/V tile.
// ---------------------------------------------------------------------------

typedef __bf16 bf16;
typedef __attribute__((ext_vector_type(16))) __bf16 bf16x16;
typedef __attribute__((ext_vector_type(8)))  __bf16 bf16x8;
typedef __attribute__((ext_vector_type(4)))  __bf16 bf16x4;
typedef __attribute__((ext_vector_type(8)))  float  f32x8;
typedef __attribute__((ext_vector_type(4)))  unsigned int u32x4;
typedef __attribute__((ext_vector_type(8)))  int    i32x8;
typedef __attribute__((ext_vector_type(4)))  int    i32x4;

#define WMMA_BF16(A, B, C) \
  __builtin_amdgcn_wmma_f32_16x16x32_bf16(false, (A), false, (B), (short)0, (C), false, false)

#if defined(__AMDGCN__) && __has_builtin(__builtin_amdgcn_tensor_load_to_lds) && \
    __has_builtin(__builtin_amdgcn_s_wait_tensorcnt)
#define HAVE_TDM 1
#endif

static constexpr int Ntok = 4096;
static constexpr int C    = 1024;
static constexpr int Hh   = 16;
static constexpr int Mffn = 4096;

__device__ __forceinline__ bf16x16 cat8(bf16x8 lo, bf16x8 hi) {
    return __builtin_shufflevector(lo, hi, 0, 1, 2, 3, 4, 5, 6, 7,
                                           8, 9, 10, 11, 12, 13, 14, 15);
}

#ifdef HAVE_TDM
// Issue one TDM 2-D tile load: global (width_elems x height, bf16, row stride
// stride_elems) -> LDS at byte offset lds_off. D# per CDNA5 ISA 8.3/8.4:
// group0 = {count=1, lds_addr, global_addr, type=2}, group1 = {mask=0,
// data_size=1(2B), tensor_dim0/1, tile_dim0/1, tensor_dim0_stride}.
__device__ __forceinline__ void tdm_load_tile_2d(unsigned lds_off, const void* gptr,
                                                 unsigned width_elems, unsigned height,
                                                 unsigned stride_elems)
{
    const unsigned long long ga = (unsigned long long)(size_t)gptr;
    u32x4 g0;
    g0[0] = 1u;                                              // count=1, user mode
    g0[1] = lds_off;                                         // lds_addr (bytes)
    g0[2] = (unsigned)ga;                                    // global_addr[31:0]
    g0[3] = (unsigned)((ga >> 32) & 0x01FFFFFFu) | (2u << 30); // addr[56:32] | type=2
    i32x8 g1;
    g1[0] = 0x00010000;                                      // data_size=1 (2 bytes)
    g1[1] = (int)(width_elems << 16);                        // tensor_dim0[15:0]
    g1[2] = (int)(height << 16);                             // tensor_dim0 hi | tensor_dim1 lo
    g1[3] = (int)(width_elems << 16);                        // tensor_dim1 hi | tile_dim0
    g1[4] = (int)height;                                     // tile_dim1 | tile_dim2=0
    g1[5] = (int)stride_elems;                               // tensor_dim0_stride[31:0]
    g1[6] = 0;                                               // stride0 hi | stride1 lo
    g1[7] = 0;
    const i32x4 z4 = {0, 0, 0, 0};
#if defined(__clang_major__) && __clang_major__ >= 23
    const i32x8 z8 = {0, 0, 0, 0, 0, 0, 0, 0};
    __builtin_amdgcn_tensor_load_to_lds(g0, g1, z4, z4, z8, 0);
#else
    __builtin_amdgcn_tensor_load_to_lds(g0, g1, z4, z4, 0);
#endif
}
#endif

// ---------------------------------------------------------------------------
// RMSNorm: one block per row (C=1024). Output bf16 = x * rsqrt(mean(x^2)+eps) * g
// ---------------------------------------------------------------------------
__global__ __launch_bounds__(256)
void rmsnorm_kernel(const float* __restrict__ x, const float* __restrict__ g,
                    bf16* __restrict__ out)
{
    const int row = blockIdx.x;
    const int t   = threadIdx.x;
    const float4 v = ((const float4*)(x + (size_t)row * C))[t];
    float ss = v.x * v.x + v.y * v.y + v.z * v.z + v.w * v.w;
    #pragma unroll
    for (int m = 1; m < 32; m <<= 1) ss += __shfl_xor(ss, m, 32);
    __shared__ float red[8];
    const int wid = t >> 5, lane = t & 31;
    if (lane == 0) red[wid] = ss;
    __syncthreads();
    float tot = 0.f;
    #pragma unroll
    for (int i = 0; i < 8; ++i) tot += red[i];
    const float inv = rsqrtf(tot * (1.0f / (float)C) + 1e-6f);
    const float4 gv = ((const float4*)g)[t];
    bf16x4 o;
    o[0] = (bf16)(v.x * inv * gv.x);
    o[1] = (bf16)(v.y * inv * gv.y);
    o[2] = (bf16)(v.z * inv * gv.z);
    o[3] = (bf16)(v.w * inv * gv.w);
    *(bf16x4*)(out + (size_t)row * C + t * 4) = o;
}

// ---------------------------------------------------------------------------
// Transpose + fp32->bf16: wt[n*K + k] = (bf16) w[k*N + n].  32x32 LDS tile.
// ---------------------------------------------------------------------------
__global__ __launch_bounds__(256)
void transconv_kernel(const float* __restrict__ w, bf16* __restrict__ wt,
                      int K, int Nn)
{
    __shared__ float tile[32][33];
    const int tx = threadIdx.x & 31;
    const int ty = threadIdx.x >> 5;           // 0..7
    const int k0 = blockIdx.x * 32;
    const int n0 = blockIdx.y * 32;
    #pragma unroll
    for (int j = 0; j < 4; ++j)
        tile[ty + j * 8][tx] = w[(size_t)(k0 + ty + j * 8) * Nn + n0 + tx];
    __syncthreads();
    #pragma unroll
    for (int j = 0; j < 4; ++j)
        wt[(size_t)(n0 + ty + j * 8) * K + k0 + tx] = (bf16)tile[tx][ty + j * 8];
}

// ---------------------------------------------------------------------------
// bf16 WMMA GEMM: C[M x N] = A[M x K] * Bt^T  (Bt stored [N x K], bf16).
// Block = 256 thr (8 waves). Block tile 128x128; wave tile 32x64 (2x4 wmma
// accumulators). Fragments follow the CDNA5 ISA VGPR layouts.
// EPI: 0 = store bf16, 1 = tanh-GELU -> bf16, 2 = +Res(fp32) -> fp32
// ---------------------------------------------------------------------------
template<int EPI>
__global__ __launch_bounds__(256)
void gemm_bf16_kernel(const bf16* __restrict__ A, const bf16* __restrict__ Bt,
                      const float* __restrict__ Res, float* __restrict__ Cf,
                      bf16* __restrict__ Cb, int Nn, int K)
{
    const int wid   = threadIdx.x >> 5;
    const int lane  = threadIdx.x & 31;
    const int l16   = lane & 15;
    const int grp   = lane >> 4;                 // 0: lanes 0-15, 1: lanes 16-31
    const int rbase = blockIdx.y * 128 + (wid >> 1) * 32;
    const int cbase = blockIdx.x * 128 + (wid & 1) * 64;

    f32x8 acc[2][4] = {};

    for (int kb = 0; kb < K; kb += 32) {
        bf16x16 a[2];
        #pragma unroll
        for (int rs = 0; rs < 2; ++rs) {
            const bf16* pa = A + (size_t)(rbase + rs * 16 + l16) * K + kb + grp * 8;
            a[rs] = cat8(*(const bf16x8*)pa, *(const bf16x8*)(pa + 16));
        }
        bf16x16 b[4];
        #pragma unroll
        for (int ct = 0; ct < 4; ++ct)
            b[ct] = cat8(*(const bf16x8*)(Bt + (size_t)(cbase + ct * 16 + l16) * K + kb + grp * 8),
                         *(const bf16x8*)(Bt + (size_t)(cbase + ct * 16 + l16) * K + kb + grp * 8 + 16));

        #pragma unroll
        for (int rs = 0; rs < 2; ++rs)
            #pragma unroll
            for (int ct = 0; ct < 4; ++ct)
                acc[rs][ct] = WMMA_BF16(a[rs], b[ct], acc[rs][ct]);

        if (kb + 32 < K) {  // speculative prefetch of next K tiles (global_prefetch_b8)
            __builtin_prefetch(A  + (size_t)(rbase + l16) * K + kb + 32, 0, 1);
            __builtin_prefetch(Bt + (size_t)(cbase + l16) * K + kb + 32, 0, 1);
        }
    }

    #pragma unroll
    for (int rs = 0; rs < 2; ++rs) {
        #pragma unroll
        for (int ct = 0; ct < 4; ++ct) {
            #pragma unroll
            for (int i = 0; i < 8; ++i) {
                const int row = rbase + rs * 16 + (grp ? i + 8 : i);
                const int col = cbase + ct * 16 + l16;
                const float v = acc[rs][ct][i];
                if constexpr (EPI == 0) {
                    Cb[(size_t)row * Nn + col] = (bf16)v;
                } else if constexpr (EPI == 1) {
                    const float u  = 0.7978845608028654f * (v + 0.044715f * v * v * v);
                    const float ge = 0.5f * v * (1.0f + tanhf(u));
                    Cb[(size_t)row * Nn + col] = (bf16)ge;
                } else {
                    Cf[(size_t)row * Nn + col] = v + Res[(size_t)row * Nn + col];
                }
            }
        }
    }
}

// ---------------------------------------------------------------------------
// Flash attention, bf16 WMMA, double-buffered TDM pipeline.
// qkv layout [N][3*C] bf16. Block = 8 waves x 16 query rows = 128 rows;
// grid = (N/128, H). Per 32-key tile: K DMA'd by TDM into ldsK[buf] one tile
// ahead of compute (fallback: register-pipelined cooperative copy); V loaded
// one tile ahead into registers and stored transposed (ldsVt[buf][d][key]).
// One barrier per tile. Online softmax in fp32.
// ---------------------------------------------------------------------------
__device__ __forceinline__ float groupMax16(float v) {
    #pragma unroll
    for (int m = 1; m < 16; m <<= 1) v = fmaxf(v, __shfl_xor(v, m, 32));
    return v;
}
__device__ __forceinline__ float groupSum16(float v) {
    #pragma unroll
    for (int m = 1; m < 16; m <<= 1) v += __shfl_xor(v, m, 32);
    return v;
}

__global__ __launch_bounds__(256)
void flash_attn_kernel(const bf16* __restrict__ qkv, bf16* __restrict__ out)
{
    __shared__ bf16 ldsK[2][32 * 64];     // row-major [key][d], double-buffered
    __shared__ bf16 ldsVt[2][64 * 32];    // transposed [d][key], double-buffered
    __shared__ bf16 ldsP[8][16 * 32];     // per-wave P scratch [qrow][key]

    const int head = blockIdx.y;
    const int wid  = threadIdx.x >> 5;
    const int lane = threadIdx.x & 31;
    const int l16  = lane & 15;
    const int grp  = lane >> 4;
    const int qrow = blockIdx.x * 128 + wid * 16;

    // Q fragments for this wave's 16 rows: two 16x32 A-fragments over D=64.
    bf16x16 qf[2];
    #pragma unroll
    for (int dh = 0; dh < 2; ++dh) {
        const bf16* pq = qkv + (size_t)(qrow + l16) * (3 * C) + head * 64 + dh * 32 + grp * 8;
        qf[dh] = cat8(*(const bf16x8*)pq, *(const bf16x8*)(pq + 16));
    }

    // Per-thread staging coordinates.
    const int vr  = threadIdx.x & 31;           // V: key row 0..31
    const int vds = (threadIdx.x >> 5) * 8;     // V: d segment 0..56
    const bf16* vbase = qkv + 2 * C + head * 64 + vds;
#ifndef HAVE_TDM
    const int kr = threadIdx.x >> 3;            // K: key row 0..31
    const int ks = (threadIdx.x & 7) * 8;       // K: d segment
    const bf16* kbase = qkv + C + head * 64 + ks;
#endif

    f32x8 o[4] = {};
    float mrun[8], lrun[8];
    #pragma unroll
    for (int i = 0; i < 8; ++i) { mrun[i] = -3.0e38f; lrun[i] = 0.f; }

    // ---- prologue: stage tile 0 into buffer 0 ----
#ifdef HAVE_TDM
    if (wid == 0)
        tdm_load_tile_2d((unsigned)(size_t)(void*)&ldsK[0][0],
                         qkv + C + head * 64, 64u, 32u, 3u * C);
#else
    *(bf16x8*)(&ldsK[0][kr * 64 + ks]) = *(const bf16x8*)(kbase + (size_t)kr * (3 * C));
#endif
    {
        const bf16x8 v0 = *(const bf16x8*)(vbase + (size_t)vr * (3 * C));
        #pragma unroll
        for (int e = 0; e < 8; ++e) ldsVt[0][(vds + e) * 32 + vr] = v0[e];
    }
#ifdef HAVE_TDM
    if (wid == 0) __builtin_amdgcn_s_wait_tensorcnt(0);
#endif
    __syncthreads();

    const int TILES = Ntok / 32;
    for (int t = 0; t < TILES; ++t) {
        const int cur  = t & 1;
        const int nxt  = cur ^ 1;
        const bool more = (t + 1) < TILES;

        // ---- issue next tile's data movement (overlaps compute below) ----
        bf16x8 vnx = {};
#ifndef HAVE_TDM
        bf16x8 knx = {};
#endif
        if (more) {
            const size_t jn = (size_t)(t + 1) * 32;
#ifdef HAVE_TDM
            if (wid == 0)
                tdm_load_tile_2d((unsigned)(size_t)(void*)&ldsK[nxt][0],
                                 qkv + jn * (3 * C) + C + head * 64, 64u, 32u, 3u * C);
#else
            knx = *(const bf16x8*)(kbase + (jn + kr) * (3 * C));
#endif
            vnx = *(const bf16x8*)(vbase + (jn + vr) * (3 * C));
        }

        // ---- compute on current tile ----
        // S = Q * K^T for two 16-key subtiles (fp32 accum).
        f32x8 s[2] = {};
        #pragma unroll
        for (int tl = 0; tl < 2; ++tl) {
            #pragma unroll
            for (int dh = 0; dh < 2; ++dh) {
                const bf16x16 kb =
                    cat8(*(const bf16x8*)(&ldsK[cur][(tl * 16 + l16) * 64 + dh * 32 + grp * 8]),
                         *(const bf16x8*)(&ldsK[cur][(tl * 16 + l16) * 64 + dh * 32 + grp * 8 + 16]));
                s[tl] = WMMA_BF16(qf[dh], kb, s[tl]);
            }
        }

        // Online softmax update; emit P (bf16) to this wave's LDS scratch.
        #pragma unroll
        for (int i = 0; i < 8; ++i) {
            const float s0 = s[0][i] * 0.125f;       // 1/sqrt(64)
            const float s1 = s[1][i] * 0.125f;
            const float tm = groupMax16(fmaxf(s0, s1));
            const float mn = fmaxf(mrun[i], tm);
            const float al = __expf(mrun[i] - mn);
            mrun[i] = mn;
            const float p0 = __expf(s0 - mn);
            const float p1 = __expf(s1 - mn);
            lrun[i] = lrun[i] * al + groupSum16(p0 + p1);
            #pragma unroll
            for (int dt = 0; dt < 4; ++dt) o[dt][i] *= al;
            const int pr = grp ? i + 8 : i;
            ldsP[wid][pr * 32 + l16]      = (bf16)p0;
            ldsP[wid][pr * 32 + 16 + l16] = (bf16)p1;
        }

        // Re-read P in A-fragment layout (same-wave LDS dependency; DS in-order).
        bf16x16 pf;
        {
            const bf16* pp = &ldsP[wid][l16 * 32 + grp * 8];
            pf = cat8(*(const bf16x8*)pp, *(const bf16x8*)(pp + 16));
        }

        // O += P * V; V B-fragment is two 16B reads from ldsVt (keys grp*8.., 16+grp*8..).
        #pragma unroll
        for (int dt = 0; dt < 4; ++dt) {
            const bf16x16 vf =
                cat8(*(const bf16x8*)(&ldsVt[cur][(dt * 16 + l16) * 32 + grp * 8]),
                     *(const bf16x8*)(&ldsVt[cur][(dt * 16 + l16) * 32 + grp * 8 + 16]));
            o[dt] = WMMA_BF16(pf, vf, o[dt]);
        }

        // ---- commit next tile's staging, drain DMA, single barrier ----
        if (more) {
#ifndef HAVE_TDM
            *(bf16x8*)(&ldsK[nxt][kr * 64 + ks]) = knx;
#endif
            #pragma unroll
            for (int e = 0; e < 8; ++e) ldsVt[nxt][(vds + e) * 32 + vr] = vnx[e];
#ifdef HAVE_TDM
            if (wid == 0) __builtin_amdgcn_s_wait_tensorcnt(0);
#endif
        }
        __syncthreads();
    }

    // Normalize and store: out[n][head*64 + d] bf16.
    #pragma unroll
    for (int dt = 0; dt < 4; ++dt) {
        #pragma unroll
        for (int i = 0; i < 8; ++i) {
            const int n   = qrow + (grp ? i + 8 : i);
            const int col = head * 64 + dt * 16 + l16;
            out[(size_t)n * C + col] = (bf16)(o[dt][i] / lrun[i]);
        }
    }
}

// ---------------------------------------------------------------------------
extern "C" void kernel_launch(void* const* d_in, const int* in_sizes, int n_in,
                              void* d_out, int out_size, void* d_ws, size_t ws_size,
                              hipStream_t stream)
{
    const float* x     = (const float*)d_in[0];
    const float* g1    = (const float*)d_in[1];
    const float* g2    = (const float*)d_in[2];
    const float* w_qkv = (const float*)d_in[3];   // [C, 3C]
    const float* w_o   = (const float*)d_in[4];   // [C, C]
    const float* w1    = (const float*)d_in[5];   // [C, M]
    const float* w2    = (const float*)d_in[6];   // [M, C]
    float* outp        = (float*)d_out;

    // Workspace carve-up (256B aligned).
    size_t off = 0;
    auto carve = [&](size_t bytes) {
        size_t r = off;
        off += (bytes + 255) & ~(size_t)255;
        return r;
    };
    char* ws = (char*)d_ws;
    bf16*  h      = (bf16*)(ws + carve((size_t)Ntok * C * 2));          // rmsnorm1 out
    bf16*  wqkvT  = (bf16*)(ws + carve((size_t)(3 * C) * C * 2));       // [3C, C]
    bf16*  woT    = (bf16*)(ws + carve((size_t)C * C * 2));             // [C, C]
    bf16*  w1T    = (bf16*)(ws + carve((size_t)Mffn * C * 2));          // [M, C]
    bf16*  w2T    = (bf16*)(ws + carve((size_t)C * Mffn * 2));          // [C, M]
    bf16*  qkvb   = (bf16*)(ws + carve((size_t)Ntok * 3 * C * 2));      // [N, 3C]
    bf16*  attnb  = (bf16*)(ws + carve((size_t)Ntok * C * 2));          // [N, C]
    float* x2     = (float*)(ws + carve((size_t)Ntok * C * 4));         // residual 1
    bf16*  h2     = (bf16*)(ws + carve((size_t)Ntok * C * 2));          // rmsnorm2 out
    bf16*  act    = (bf16*)(ws + carve((size_t)Ntok * Mffn * 2));       // gelu(h2@w1)
    (void)ws_size; (void)in_sizes; (void)n_in; (void)out_size;

    // 1) h = rmsnorm(x, g1) -> bf16
    rmsnorm_kernel<<<Ntok, 256, 0, stream>>>(x, g1, h);

    // 2) weight transpose+convert to bf16 [N,K]
    transconv_kernel<<<dim3(C / 32, (3 * C) / 32), 256, 0, stream>>>(w_qkv, wqkvT, C, 3 * C);
    transconv_kernel<<<dim3(C / 32, C / 32),       256, 0, stream>>>(w_o,   woT,   C, C);
    transconv_kernel<<<dim3(C / 32, Mffn / 32),    256, 0, stream>>>(w1,    w1T,   C, Mffn);
    transconv_kernel<<<dim3(Mffn / 32, C / 32),    256, 0, stream>>>(w2,    w2T,   Mffn, C);

    // 3) qkv = h @ w_qkv  -> bf16 [N, 3C]
    gemm_bf16_kernel<0><<<dim3((3 * C) / 128, Ntok / 128), 256, 0, stream>>>(
        h, wqkvT, nullptr, nullptr, qkvb, 3 * C, C);

    // 4) flash attention -> attnb bf16 [N, C]
    flash_attn_kernel<<<dim3(Ntok / 128, Hh), 256, 0, stream>>>(qkvb, attnb);

    // 5) x2 = x + attnb @ w_o   (fp32)
    gemm_bf16_kernel<2><<<dim3(C / 128, Ntok / 128), 256, 0, stream>>>(
        attnb, woT, x, x2, nullptr, C, C);

    // 6) h2 = rmsnorm(x2, g2) -> bf16
    rmsnorm_kernel<<<Ntok, 256, 0, stream>>>(x2, g2, h2);

    // 7) act = gelu(h2 @ w1) -> bf16 [N, M]
    gemm_bf16_kernel<1><<<dim3(Mffn / 128, Ntok / 128), 256, 0, stream>>>(
        h2, w1T, nullptr, nullptr, act, Mffn, C);

    // 8) out = x2 + act @ w2  (fp32)
    gemm_bf16_kernel<2><<<dim3(C / 128, Ntok / 128), 256, 0, stream>>>(
        act, w2T, x2, outp, nullptr, C, Mffn);
}
